// BiLSTM_CRF_33483565039758
// MI455X (gfx1250) — matrix operand order repaired and checked by reading, not compiled, not run on hardware
//
#include <hip/hip_runtime.h>
#include <hip/hip_bf16.h>
#include <math.h>

// ---------------------------------------------------------------------------
// BiLSTM-CRF for MI455X (gfx1250): bf16 WMMA (16x16x32) for all matmuls,
// TDM (tensor_load_to_lds) staging of GEMM operands and per-step LSTM
// preactivations into LDS, persistent-workgroup LSTM scan, f32 Viterbi.
// ---------------------------------------------------------------------------

#define BATCH 32
#define TLEN  512
#define EDIM  512      // embedding dim == 2H
#define HDIM  256
#define FOURH 1024
#define NLAY  2
#define NLB   20       // labels
#define MTOT  (BATCH * TLEN)   // 16384 tokens

typedef __attribute__((ext_vector_type(16))) __bf16 v16bf;
typedef __attribute__((ext_vector_type(8)))  __bf16 v8bf;
typedef __attribute__((ext_vector_type(8)))  float  v8f;
typedef __attribute__((ext_vector_type(4)))  unsigned int u32x4;
typedef __attribute__((ext_vector_type(8)))  unsigned int u32x8;

// ---------------------------------------------------------------------------
// TDM 2D tile load: global (row-major, bf16/2-byte elements) -> LDS.
// Builds a Tensor DMA Descriptor (D#) per CDNA5 ISA ch.8 and issues
// tensor_load_to_lds with SGPR groups 0 (4 SGPRs) and 1 (8 SGPRs); groups
// 2/3 are NULL (2D tensor). Tracked with TENSORcnt.
//   gaddr        : global byte address of tile start
//   lds_off      : LDS byte offset of destination
//   tensor_d0/d1 : tensor extent (elements / lines) for OOB handling
//   tile_d0/d1   : tile extent (elements / lines)
//   stride0      : elements between consecutive dim-1 lines
// ---------------------------------------------------------------------------
__device__ __forceinline__ void tdm_load_2d_bf16(const void* gaddr, unsigned lds_off,
                                                 unsigned tensor_d0, unsigned tensor_d1,
                                                 unsigned tile_d0, unsigned tile_d1,
                                                 unsigned long long stride0) {
  unsigned long long ga = (unsigned long long)gaddr;
  u32x4 g0;
  g0[0] = 1u;                                        // count=1, user mode, no gather
  g0[1] = lds_off;                                   // lds_addr
  g0[2] = (unsigned)(ga & 0xffffffffu);              // global_addr[31:0]
  g0[3] = (unsigned)((ga >> 32) & 0x01ffffffu)       // global_addr[56:32]
          | (2u << 30);                              // type = 2 ("image")
  u32x8 g1;
  g1[0] = 1u << 16;                                  // data_size=1 (2 bytes)
  g1[1] = (tensor_d0 & 0xffffu) << 16;               // tensor_dim0[15:0]
  g1[2] = (tensor_d0 >> 16) | ((tensor_d1 & 0xffffu) << 16);
  g1[3] = (tensor_d1 >> 16) | ((tile_d0 & 0xffffu) << 16);   // tile_dim0
  g1[4] = tile_d1 & 0xffffu;                         // tile_dim1, tile_dim2=0
  g1[5] = (unsigned)(stride0 & 0xffffffffu);         // tensor_dim0_stride[31:0]
  g1[6] = (unsigned)((stride0 >> 32) & 0xffffu);     // stride[47:32], dim1_stride=0
  g1[7] = 0u;
  asm volatile("tensor_load_to_lds %0, %1" :: "s"(g0), "s"(g1) : "memory");
}

// --- A-operand loader: 16x32 bf16 tile, row-major source with leading dim ld.
// Contiguous k order (matches B loads): lanes 0-15 hold K[0..15], lanes 16-31 K[16..31].
__device__ __forceinline__ v16bf load_a_tile(const __bf16* base, int ld, int row,
                                             int k0, int hf) {
  const __bf16* p = base + (size_t)row * ld + k0 + hf * 16;
  v8bf lo = *(const v8bf*)p;
  v8bf hi = *(const v8bf*)(p + 8);
  v16bf r;
#pragma unroll
  for (int i = 0; i < 8; ++i) { r[i] = lo[i]; r[i + 8] = hi[i]; }
  return r;
}

// ---------------------------------------------------------------------------
// Embedding gather: x_bf16[b*T+t][0..511] = table[input[b*T+t]]
// ---------------------------------------------------------------------------
__global__ void embed_kernel(const int* __restrict__ inp,
                             const float* __restrict__ table,
                             __bf16* __restrict__ xout) {
  int row = blockIdx.x;
  int tok = inp[row];
  const float* src = table + (size_t)tok * EDIM;
  __bf16* dst = xout + (size_t)row * EDIM;
  for (int i = threadIdx.x; i < EDIM; i += blockDim.x)
    dst[i] = (__bf16)src[i];
}

__global__ void cvt_bf16(const float* __restrict__ src, __bf16* __restrict__ dst, int n) {
  int i = blockIdx.x * blockDim.x + threadIdx.x;
  if (i < n) dst[i] = (__bf16)src[i];
}

// lin_W [20][512] -> bf16 padded to [32][512] (zeros for rows >= 20)
__global__ void cvt_linw(const float* __restrict__ src, __bf16* __restrict__ dst) {
  int i = blockIdx.x * blockDim.x + threadIdx.x;
  if (i < 32 * EDIM)
    dst[i] = (i < NLB * EDIM) ? (__bf16)src[i] : (__bf16)0.0f;
}

// ---------------------------------------------------------------------------
// Input-projection GEMM with TDM-staged operands.
//   D_bf16[M,2048] = A[M,512](bf16) @ W[2048,512](bf16)^T + bias
// 8 waves/block; block covers one A stripe (16 rows x K) and 8 N-tiles
// (128 W rows x K). Wave 0 TDM-loads W chunk (128KB) + A chunk (16KB) into
// LDS, waits TENSORcnt, barrier, then all waves run ds_load -> v_wmma.
// ---------------------------------------------------------------------------
#define PRE_N   (2 * FOURH)              // 2048
#define GEMM_LDS_W 0
#define GEMM_LDS_A (128 * EDIM * 2)      // 131072
#define GEMM_LDS_BYTES (GEMM_LDS_A + 16 * EDIM * 2)  // 147456

__global__ void gemm_pre_tdm(const __bf16* __restrict__ A, const __bf16* __restrict__ W,
                             const float* __restrict__ bias,
                             __bf16* __restrict__ D) {
  extern __shared__ char smem[];
  __bf16* bS = (__bf16*)(smem + GEMM_LDS_W);   // [128][512] W rows
  __bf16* aS = (__bf16*)(smem + GEMM_LDS_A);   // [16][512]  A rows

  const int lane = threadIdx.x & 31;
  const int wave = threadIdx.x >> 5;
  const int hf   = lane >> 4;
  const int ntiles = PRE_N >> 4;               // 128
  const int mt = (blockIdx.x * 8) / ntiles;    // all 8 waves share mt
  const int n0 = ((blockIdx.x * 8) % ntiles) * 16;   // first of 128 W rows

  if (wave == 0) {
    unsigned ldsBase = (unsigned)(unsigned long long)(uintptr_t)smem;
    tdm_load_2d_bf16(W + (size_t)n0 * EDIM, ldsBase + GEMM_LDS_W,
                     EDIM, 128, EDIM, 128, EDIM);
    tdm_load_2d_bf16(A + (size_t)mt * 16 * EDIM, ldsBase + GEMM_LDS_A,
                     EDIM, 16, EDIM, 16, EDIM);
    __builtin_amdgcn_s_wait_tensorcnt((short)0);
  }
  __syncthreads();

  const int col = n0 + wave * 16 + (lane & 15);      // global N column
  const int lrow = wave * 16 + (lane & 15);          // W row within LDS chunk

  v8f acc;
  const float bv = bias[col];
#pragma unroll
  for (int e = 0; e < 8; ++e) acc[e] = bv;

#pragma unroll
  for (int k0 = 0; k0 < EDIM; k0 += 32) {
    v16bf a = load_a_tile(aS, EDIM, lane & 15, k0, hf);
    v16bf b = *(const v16bf*)(bS + (size_t)lrow * EDIM + k0 + hf * 16);
    acc = __builtin_amdgcn_wmma_f32_16x16x32_bf16(
        false, a, false, b, (short)0, acc, false, false);
  }

#pragma unroll
  for (int e = 0; e < 8; ++e) {
    int m = mt * 16 + hf * 8 + e;
    D[(size_t)m * PRE_N + col] = (__bf16)acc[e];
  }
}

// ---------------------------------------------------------------------------
// Head GEMM (tiny): em[M,20] = A[M,512] @ linW[32,512]^T + lin_b, f32 store.
// ---------------------------------------------------------------------------
__global__ void gemm_head(const __bf16* __restrict__ A, const __bf16* __restrict__ W,
                          const float* __restrict__ bias, float* __restrict__ Df) {
  const int lane = threadIdx.x & 31;
  const int wid  = blockIdx.x * (blockDim.x >> 5) + (threadIdx.x >> 5);
  const int ntiles = 2;                         // N = 32
  const int mt = wid / ntiles, nt = wid % ntiles;
  const int col = nt * 16 + (lane & 15);
  const int hf  = lane >> 4;

  v8f acc;
  const float bv = (col < NLB) ? bias[col] : 0.0f;
#pragma unroll
  for (int e = 0; e < 8; ++e) acc[e] = bv;

  const __bf16* wrow = W + (size_t)col * EDIM;
  for (int k0 = 0; k0 < EDIM; k0 += 32) {
    v16bf a = load_a_tile(A, EDIM, mt * 16 + (lane & 15), k0, hf);
    v16bf b = *(const v16bf*)(wrow + k0 + hf * 16);
    if (k0 + 64 < EDIM) __builtin_prefetch(wrow + k0 + 64, 0, 0);
    acc = __builtin_amdgcn_wmma_f32_16x16x32_bf16(
        false, a, false, b, (short)0, acc, false, false);
  }
  if (col < NLB) {
#pragma unroll
    for (int e = 0; e < 8; ++e) {
      int m = mt * 16 + hf * 8 + e;
      Df[(size_t)m * NLB + col] = acc[e];
    }
  }
}

// ---------------------------------------------------------------------------
// Recurrent LSTM scan. grid = 2 (blockIdx.x = direction). 512 thr = 16 waves.
// Per step: g = pre[t] + h @ Whh^T (128 WMMA tiles, 8/wave, K=256), then
// gate nonlinearities update c (f32 LDS) / h (bf16 LDS) and emit h.
// The 64KB pre-activation slab for step t+1 is TDM-loaded into a double
// buffer while step t computes (one tensor_load_to_lds per step, hidden
// behind ~1024 WMMAs).  LDS: h 16K + c 32K + g 128K + 2x pre 64K = 304 KB.
// ---------------------------------------------------------------------------
#define REC_LDS_H   0
#define REC_LDS_C   (BATCH * HDIM * 2)                    // 16384
#define REC_LDS_G   (REC_LDS_C + BATCH * HDIM * 4)        // 49152
#define REC_LDS_PRE (REC_LDS_G + BATCH * FOURH * 4)       // 180224
#define REC_PRE_BYTES (BATCH * FOURH * 2)                 // 65536 per buffer
#define REC_LDS_BYTES (REC_LDS_PRE + 2 * REC_PRE_BYTES)   // 311296

__global__ void lstm_recurrent(const __bf16* __restrict__ pre,   // [MTOT][2048]
                               const __bf16* __restrict__ whh,   // [2][1024][256]
                               __bf16* __restrict__ out) {       // [MTOT][512]
  extern __shared__ char smem[];
  __bf16* hS   = (__bf16*)(smem + REC_LDS_H);
  float*  cS   = (float*)(smem + REC_LDS_C);
  float*  gS   = (float*)(smem + REC_LDS_G);
  __bf16* preS = (__bf16*)(smem + REC_LDS_PRE);   // [2][BATCH][FOURH]

  const int dir  = blockIdx.x;
  const int tid  = threadIdx.x;
  const int lane = tid & 31;
  const int wave = tid >> 5;
  const int hf   = lane >> 4;
  const bool rev = (dir == 1);
  const __bf16* whhd = whh + (size_t)dir * FOURH * HDIM;
  const int preofs = dir * FOURH;
  const int outofs = dir * HDIM;
  const unsigned ldsBase = (unsigned)(unsigned long long)(uintptr_t)smem;

  for (int i = tid; i < BATCH * HDIM; i += blockDim.x) {
    hS[i] = (__bf16)0.0f;
    cS[i] = 0.0f;
  }
  // prefetch pre slab for t=0 into buffer 0: rows m=0..31 at stride T*2048
  if (wave == 0) {
    int tt0 = rev ? (TLEN - 1) : 0;
    tdm_load_2d_bf16(pre + (size_t)tt0 * PRE_N + preofs, ldsBase + REC_LDS_PRE,
                     FOURH, BATCH, FOURH, BATCH, (unsigned long long)TLEN * PRE_N);
    __builtin_amdgcn_s_wait_tensorcnt((short)0);
  }
  __syncthreads();

  for (int t = 0; t < TLEN; ++t) {
    const int tt = rev ? (TLEN - 1 - t) : t;
    const __bf16* preCur = preS + (size_t)(t & 1) * BATCH * FOURH;

    // issue TDM for step t+1 into the other buffer (overlaps with compute)
    if (wave == 0 && t + 1 < TLEN) {
      int ttn = rev ? (TLEN - 2 - t) : (t + 1);
      tdm_load_2d_bf16(pre + (size_t)ttn * PRE_N + preofs,
                       ldsBase + REC_LDS_PRE + ((t + 1) & 1) * REC_PRE_BYTES,
                       FOURH, BATCH, FOURH, BATCH, (unsigned long long)TLEN * PRE_N);
    }

    // --- g = pre[:, tt, :] + h @ Whh^T  (WMMA, acc init from LDS pre) ---
#pragma unroll
    for (int i = 0; i < 8; ++i) {
      const int tile = wave * 8 + i;
      const int mt   = tile >> 6;     // 0..1  (batch tile)
      const int nt   = tile & 63;     // 0..63 (gate tile)
      const int col  = nt * 16 + (lane & 15);

      v8f acc;
#pragma unroll
      for (int e = 0; e < 8; ++e) {
        int m = mt * 16 + hf * 8 + e;
        acc[e] = (float)preCur[m * FOURH + col];
      }
      const __bf16* wrow = whhd + (size_t)col * HDIM;
#pragma unroll
      for (int k0 = 0; k0 < HDIM; k0 += 32) {
        v16bf a = load_a_tile(hS, HDIM, mt * 16 + (lane & 15), k0, hf);
        v16bf b = *(const v16bf*)(wrow + k0 + hf * 16);
        acc = __builtin_amdgcn_wmma_f32_16x16x32_bf16(
            false, a, false, b, (short)0, acc, false, false);
      }
#pragma unroll
      for (int e = 0; e < 8; ++e) {
        int m = mt * 16 + hf * 8 + e;
        gS[m * FOURH + col] = acc[e];
      }
    }
    __syncthreads();

    // --- gates: i,f,g,o -> c,h (8192 cells / 512 threads) ---
    for (int e = tid; e < BATCH * HDIM; e += blockDim.x) {
      const int bb = e >> 8;          // batch
      const int u  = e & 255;         // hidden unit
      float gi = gS[bb * FOURH + u];
      float gf = gS[bb * FOURH + HDIM + u];
      float gg = gS[bb * FOURH + 2 * HDIM + u];
      float go = gS[bb * FOURH + 3 * HDIM + u];
      gi = 1.0f / (1.0f + __expf(-gi));
      gf = 1.0f / (1.0f + __expf(-gf));
      gg = tanhf(gg);
      go = 1.0f / (1.0f + __expf(-go));
      float c = gf * cS[e] + gi * gg;
      float h = go * tanhf(c);
      cS[e] = c;
      hS[e] = (__bf16)h;
      out[(size_t)(bb * TLEN + tt) * EDIM + outofs + u] = (__bf16)h;
    }
    // make sure next step's pre slab has landed before anyone reads it
    if (wave == 0 && t + 1 < TLEN)
      __builtin_amdgcn_s_wait_tensorcnt((short)0);
    __syncthreads();
  }
}

// ---------------------------------------------------------------------------
// Viterbi decode + backtrack. One block, 640 threads = (b,j) pairs.
// ---------------------------------------------------------------------------
__global__ void viterbi_kernel(const float* __restrict__ em,        // [MTOT][20]
                               const unsigned char* __restrict__ mask,  // [B][T]
                               const float* __restrict__ trans,     // [20][20]
                               const float* __restrict__ start,
                               const float* __restrict__ endt,
                               int* __restrict__ hist,              // [(T-1)][B][20]
                               int* __restrict__ tags) {            // [B][T]
  __shared__ float sc[BATCH * NLB];
  __shared__ float tr[NLB * NLB];
  __shared__ int   lastS[BATCH];

  const int tid = threadIdx.x;
  for (int i = tid; i < NLB * NLB; i += blockDim.x) tr[i] = trans[i];
  const int b = tid / NLB;
  const int j = tid % NLB;
  sc[b * NLB + j] = start[j] + em[(size_t)(b * TLEN) * NLB + j];
  __syncthreads();

  for (int t = 1; t < TLEN; ++t) {
    float cur = sc[b * NLB + j];
    float best = -INFINITY;
    int bi = 0;
#pragma unroll
    for (int i = 0; i < NLB; ++i) {
      float v = sc[b * NLB + i] + tr[i * NLB + j];
      if (v > best) { best = v; bi = i; }
    }
    best += em[(size_t)(b * TLEN + t) * NLB + j];
    __syncthreads();
    const bool m = mask[b * TLEN + t] != 0;
    sc[b * NLB + j] = m ? best : cur;
    hist[(size_t)(t - 1) * BATCH * NLB + b * NLB + j] = m ? bi : j;
    __syncthreads();
  }

  if (j == 0) {
    float best = -INFINITY;
    int bi = 0;
    for (int i = 0; i < NLB; ++i) {
      float v = sc[b * NLB + i] + endt[i];
      if (v > best) { best = v; bi = i; }
    }
    lastS[b] = bi;
  }
  __syncthreads();

  if (tid < BATCH) {
    int tg = lastS[tid];
    tags[tid * TLEN + TLEN - 1] = tg;
    for (int t = TLEN - 1; t >= 1; --t) {
      tg = hist[(size_t)(t - 1) * BATCH * NLB + tid * NLB + tg];
      tags[tid * TLEN + t - 1] = tg;
    }
  }
}

// ---------------------------------------------------------------------------
// Host orchestration
// ---------------------------------------------------------------------------
extern "C" void kernel_launch(void* const* d_in, const int* in_sizes, int n_in,
                              void* d_out, int out_size, void* d_ws, size_t ws_size,
                              hipStream_t stream) {
  const int*   input   = (const int*)d_in[0];
  const unsigned char* mask = (const unsigned char*)d_in[1];
  const float* table   = (const float*)d_in[2];
  const float* Wih     = (const float*)d_in[3];
  const float* Whh     = (const float*)d_in[4];
  const float* bias    = (const float*)d_in[5];
  const float* linW    = (const float*)d_in[6];
  const float* linB    = (const float*)d_in[7];
  const float* trans   = (const float*)d_in[8];
  const float* startT  = (const float*)d_in[9];
  const float* endT    = (const float*)d_in[10];
  int* tags = (int*)d_out;

  // workspace carve-up
  char* ws = (char*)d_ws;
  size_t off = 0;
  __bf16* xa = (__bf16*)(ws + off);      off += (size_t)MTOT * EDIM * 2;          // 16 MB
  __bf16* xb = (__bf16*)(ws + off);      off += (size_t)MTOT * EDIM * 2;          // 16 MB
  __bf16* wihB = (__bf16*)(ws + off);    off += (size_t)NLAY * 2 * FOURH * EDIM * 2;   // 4 MB
  __bf16* whhB = (__bf16*)(ws + off);    off += (size_t)NLAY * 2 * FOURH * HDIM * 2;   // 2 MB
  __bf16* linwB = (__bf16*)(ws + off);   off += (size_t)32 * EDIM * 2;
  __bf16* pre = (__bf16*)(ws + off);     off += (size_t)MTOT * PRE_N * 2;         // 64 MB
  float*  em  = (float*)(ws + off);      off += (size_t)MTOT * NLB * 4;
  int*    hist = (int*)(ws + off);       off += (size_t)TLEN * BATCH * NLB * 4;

  // 1. embedding -> bf16
  embed_kernel<<<MTOT, 128, 0, stream>>>(input, table, xa);

  // 2. weight conversions
  {
    int n = NLAY * 2 * FOURH * EDIM;
    cvt_bf16<<<(n + 255) / 256, 256, 0, stream>>>(Wih, wihB, n);
    n = NLAY * 2 * FOURH * HDIM;
    cvt_bf16<<<(n + 255) / 256, 256, 0, stream>>>(Whh, whhB, n);
    cvt_linw<<<(32 * EDIM + 255) / 256, 256, 0, stream>>>(linW, linwB);
  }

  // raise dynamic-LDS caps (WGP has 320 KB on CDNA5)
  hipFuncSetAttribute(reinterpret_cast<const void*>(gemm_pre_tdm),
                      hipFuncAttributeMaxDynamicSharedMemorySize, GEMM_LDS_BYTES);
  hipFuncSetAttribute(reinterpret_cast<const void*>(lstm_recurrent),
                      hipFuncAttributeMaxDynamicSharedMemorySize, REC_LDS_BYTES);

  // 3. BiLSTM layers
  __bf16* cur = xa;
  __bf16* nxt = xb;
  for (int l = 0; l < NLAY; ++l) {
    // fused both-direction input projection: [16384,512] @ [512,2048]^T + bias
    const int tiles = (MTOT / 16) * (PRE_N / 16);       // 131072
    gemm_pre_tdm<<<tiles / 8, 256, GEMM_LDS_BYTES, stream>>>(
        cur, wihB + (size_t)l * 2 * FOURH * EDIM, bias + (size_t)l * 2 * FOURH, pre);

    // sequential scan, one workgroup per direction, TDM double-buffered pre
    lstm_recurrent<<<2, 512, REC_LDS_BYTES, stream>>>(
        pre, whhB + (size_t)l * 2 * FOURH * HDIM, nxt);

    __bf16* tmp = cur; cur = nxt; nxt = tmp;
  }

  // 4. linear head: [16384,512] @ [512,32(pad)]^T, store f32 cols < 20
  {
    const int tiles = (MTOT / 16) * 2;                  // 2048
    gemm_head<<<tiles / 8, 256, 0, stream>>>(cur, linwB, linB, em);
  }

  // 5. CRF Viterbi decode + backtrack -> int32 tags [B][T]
  viterbi_kernel<<<1, BATCH * NLB, 0, stream>>>(em, mask, trans, startT, endT,
                                                hist, tags);
}
